// GraphAttentionLayer_51556787421845
// MI455X (gfx1250) — matrix operand order repaired and checked by reading, not compiled, not run on hardware
//
#include <hip/hip_runtime.h>

#define N_NODES 4096
#define DMODEL  512
#define DFF     512
#define NHEAD   4
#define ALPHA   0.3f

typedef __attribute__((ext_vector_type(16))) __bf16 v16bf;
typedef __attribute__((ext_vector_type(8)))  float  v8f;
typedef __attribute__((ext_vector_type(4)))  float  f32x4;

union Frag16 { v16bf v; f32x4 q[2]; __bf16 e[16]; };  // 32B: one WMMA A/B bf16 fragment
union Half8  { f32x4 q;  __bf16 e[8]; };              // 16B: packed bf16 store

static __device__ __forceinline__ v8f wmma_bf16(const Frag16& a, const Frag16& b, v8f c) {
  return __builtin_amdgcn_wmma_f32_16x16x32_bf16(false, a.v, false, b.v, (short)0, c,
                                                 false, false);
}

// ---------------------------------------------------------------------------
// Prep: bf16 conversions.  el = edges + labels.
// ---------------------------------------------------------------------------
__global__ __launch_bounds__(256) void gat_prep_inputs(
    const float* __restrict__ nodes, const float* __restrict__ edges,
    const float* __restrict__ labels, __bf16* __restrict__ nodes16,
    __bf16* __restrict__ el16, int total) {
  int i = blockIdx.x * blockDim.x + threadIdx.x;
  if (i < total) {
    nodes16[i] = (__bf16)nodes[i];
    el16[i]    = (__bf16)(edges[i] + labels[i]);
  }
}

// Transpose-convert weights: WeT[f][d] = We[d][f]; WT[h][f][d] = W[h][d][f]
__global__ __launch_bounds__(256) void gat_prep_weights(
    const float* __restrict__ We, const float* __restrict__ W,
    __bf16* __restrict__ WeT, __bf16* __restrict__ WT, int total /* = F*D */) {
  int i = blockIdx.x * blockDim.x + threadIdx.x;
  if (i >= total) return;
  int f = i / DMODEL, d = i - f * DMODEL;
  WeT[i] = (__bf16)We[d * DFF + f];
#pragma unroll
  for (int h = 0; h < NHEAD; ++h)
    WT[(size_t)h * DFF * DMODEL + i] = (__bf16)W[((size_t)h * DMODEL + d) * DFF + f];
}

// ---------------------------------------------------------------------------
// Fused projections (WMMA bf16):
//   E      = el @ WeT^T + be
//   feats  = nodes @ WT[h]^T + b[h]          (fp32 accum)
//   a_self/a_neigh += <feats, u/v>           (half-wave shuffle + fp32 atomics)
//   featsET[h][f][n] = bf16(feats + E)       (transposed for B-fragments)
// Block: 256 thr = 8 waves, 2 wave-rows x 4 wave-cols -> 32x64 tile.
// ---------------------------------------------------------------------------
__global__ __launch_bounds__(256) void gat_proj_kernel(
    const __bf16* __restrict__ nodes16, const __bf16* __restrict__ el16,
    const __bf16* __restrict__ WeT, const __bf16* __restrict__ WT,
    const float* __restrict__ be, const float* __restrict__ bb,
    const float* __restrict__ u, const float* __restrict__ v,
    __bf16* __restrict__ featsET, float* __restrict__ a_self,
    float* __restrict__ a_neigh) {
  const int lane = threadIdx.x & 31;
  const int wid  = threadIdx.x >> 5;
  const int wr = wid >> 2, wc = wid & 3;
  const int r = lane & 15, kh = lane >> 4;
  const int n0 = blockIdx.x * 32 + wr * 16;
  const int f0 = blockIdx.y * 64 + wc * 16;

  v8f zero;
#pragma unroll
  for (int j = 0; j < 8; ++j) zero[j] = 0.0f;

  // ---- E tile: el @ We ----
  v8f eacc = zero;
  for (int k0 = 0; k0 < DMODEL; k0 += 32) {
    Frag16 A, B;
    const __bf16* ap = el16 + (size_t)(n0 + r) * DMODEL + k0 + kh * 8;
    A.q[0] = *(const f32x4*)ap;
    A.q[1] = *(const f32x4*)(ap + 16);
    const __bf16* bp = WeT + (size_t)(f0 + r) * DMODEL + k0 + kh * 8;
    B.q[0] = *(const f32x4*)bp;
    B.q[1] = *(const f32x4*)(bp + 16);
    eacc = wmma_bf16(A, B, eacc);
  }
  const float bev = be[f0 + r];
#pragma unroll
  for (int i = 0; i < 8; ++i) eacc[i] += bev;

  // ---- per-head feats ----
  for (int h = 0; h < NHEAD; ++h) {
    v8f facc = zero;
    const __bf16* wt = WT + (size_t)h * DFF * DMODEL;
    for (int k0 = 0; k0 < DMODEL; k0 += 32) {
      Frag16 A, B;
      const __bf16* ap = nodes16 + (size_t)(n0 + r) * DMODEL + k0 + kh * 8;
      A.q[0] = *(const f32x4*)ap;
      A.q[1] = *(const f32x4*)(ap + 16);
      const __bf16* bp = wt + (size_t)(f0 + r) * DMODEL + k0 + kh * 8;
      B.q[0] = *(const f32x4*)bp;
      B.q[1] = *(const f32x4*)(bp + 16);
      facc = wmma_bf16(A, B, facc);
    }
    const float bias = bb[h * DFF + f0 + r];
    const float uu   = u[h * DFF + f0 + r];
    const float vv   = v[h * DFF + f0 + r];
    Half8 fe;
#pragma unroll
    for (int i = 0; i < 8; ++i) {
      float fb = facc[i] + bias;       // feats (pre-edge) -> used for attention dots
      float su = fb * uu, sv = fb * vv;
#pragma unroll
      for (int msk = 1; msk < 16; msk <<= 1) {  // sum over the 16 cols of this tile
        su += __shfl_xor(su, msk, 32);
        sv += __shfl_xor(sv, msk, 32);
      }
      if (r == 0) {
        atomicAdd(&a_self[h * N_NODES + n0 + kh * 8 + i], su);
        atomicAdd(&a_neigh[h * N_NODES + n0 + kh * 8 + i], sv);
      }
      fe.e[i] = (__bf16)(fb + eacc[i]);  // featsE = feats + e
    }
    // rows n0+kh*8 .. +7 are contiguous in the transposed layout -> one b128 store
    *(f32x4*)(featsET + (size_t)(h * DFF + f0 + r) * N_NODES + n0 + kh * 8) = fe.q;
  }
}

// ---------------------------------------------------------------------------
// Per-head rank-1 softmax stats:  M+ = max_m max(t,at),  m- = min_m min(t,at)
// ---------------------------------------------------------------------------
__global__ __launch_bounds__(256) void gat_stats_kernel(
    const float* __restrict__ a_neigh, float* __restrict__ stats) {
  __shared__ float smax[256], smin[256];
  const int h = blockIdx.x, tid = threadIdx.x;
  float gp = -3.4e38f, gm = 3.4e38f;
  for (int m = tid; m < N_NODES; m += 256) {
    float t = a_neigh[h * N_NODES + m];
    gp = fmaxf(gp, fmaxf(t, ALPHA * t));
    gm = fminf(gm, fminf(t, ALPHA * t));
  }
  smax[tid] = gp; smin[tid] = gm;
  __syncthreads();
  for (int s = 128; s > 0; s >>= 1) {
    if (tid < s) {
      smax[tid] = fmaxf(smax[tid], smax[tid + s]);
      smin[tid] = fminf(smin[tid], smin[tid + s]);
    }
    __syncthreads();
  }
  if (tid == 0) { stats[2 * h] = smax[0]; stats[2 * h + 1] = smin[0]; }
}

static __device__ __forceinline__ float pweight(float t, float s, float rowmax, bool pos) {
  float g = pos ? fmaxf(t, ALPHA * t) : fminf(t, ALPHA * t);
  return __expf(s * g - rowmax);  // always <= 0 in the exponent: no overflow
}

// ---------------------------------------------------------------------------
// Attention: out[n,f] = relu( mean_h (P_h @ featsE_h)[n,f] / rowsum_h[n] )
// Block: 256 thr = 8 waves; wave owns 16 rows x 128 cols (8 WMMA accumulators),
// streams all m in K=32 chunks, builds P fragment with v_exp_f32, and gets the
// row sums from a 9th WMMA against an all-ones B fragment.
// ---------------------------------------------------------------------------
__global__ __launch_bounds__(256) void gat_attn_kernel(
    const float* __restrict__ a_self, const float* __restrict__ a_neigh,
    const float* __restrict__ stats, const __bf16* __restrict__ featsET,
    float* __restrict__ out) {
  const int lane = threadIdx.x & 31;
  const int wid  = threadIdx.x >> 5;
  const int wr = wid >> 2, wc = wid & 3;
  const int r = lane & 15, kh = lane >> 4;
  const int n0 = blockIdx.x * 32 + wr * 16;

  v8f zero;
#pragma unroll
  for (int j = 0; j < 8; ++j) zero[j] = 0.0f;

  Frag16 bones;
#pragma unroll
  for (int j = 0; j < 16; ++j) bones.e[j] = (__bf16)1.0f;

  v8f tot[8];
#pragma unroll
  for (int s8 = 0; s8 < 8; ++s8) tot[s8] = zero;

  for (int h = 0; h < NHEAD; ++h) {
    const float s  = a_self[h * N_NODES + n0 + r];
    const float Mp = stats[2 * h], mmn = stats[2 * h + 1];
    const bool pos = (s >= 0.0f);
    const float rowmax = pos ? s * Mp : s * mmn;  // exact row max of lrelu(s*t)

    v8f acc[8];
#pragma unroll
    for (int s8 = 0; s8 < 8; ++s8) acc[s8] = zero;
    v8f rowacc = zero;

    const float*  tp    = a_neigh + h * N_NODES;
    const __bf16* fbase = featsET + (size_t)h * DFF * N_NODES;

    for (int m0 = 0; m0 < N_NODES; m0 += 32) {
      const float* tb = tp + m0 + kh * 8;
      f32x4 t0 = *(const f32x4*)(tb);
      f32x4 t1 = *(const f32x4*)(tb + 4);
      f32x4 t2 = *(const f32x4*)(tb + 16);
      f32x4 t3 = *(const f32x4*)(tb + 20);
      Frag16 A;  // K = (j&7) + 16*(j>>3) + 8*kh
#pragma unroll
      for (int j = 0; j < 4; ++j) {
        A.e[j]      = (__bf16)pweight(t0[j], s, rowmax, pos);
        A.e[j + 4]  = (__bf16)pweight(t1[j], s, rowmax, pos);
        A.e[j + 8]  = (__bf16)pweight(t2[j], s, rowmax, pos);
        A.e[j + 12] = (__bf16)pweight(t3[j], s, rowmax, pos);
      }
      const __bf16* bp0 = fbase + m0 + kh * 8;
#pragma unroll
      for (int sub = 0; sub < 8; ++sub) {
        const __bf16* bp = bp0 + (size_t)(wc * 128 + sub * 16 + r) * N_NODES;
        Frag16 B;
        B.q[0] = *(const f32x4*)bp;
        B.q[1] = *(const f32x4*)(bp + 16);
        acc[sub] = wmma_bf16(A, B, acc[sub]);
      }
      rowacc = wmma_bf16(A, bones, rowacc);  // rowsum, aligned with acc rows
    }
#pragma unroll
    for (int i = 0; i < 8; ++i) {
      float inv = 1.0f / rowacc[i];  // rowsum >= 1 (max element gives exp(0)=1)
#pragma unroll
      for (int sub = 0; sub < 8; ++sub) tot[sub][i] += acc[sub][i] * inv;
    }
  }
#pragma unroll
  for (int i = 0; i < 8; ++i) {
    float* orow = out + (size_t)(n0 + kh * 8 + i) * DFF + wc * 128 + r;
#pragma unroll
    for (int sub = 0; sub < 8; ++sub)
      orow[sub * 16] = fmaxf(tot[sub][i] * 0.25f, 0.0f);  // mean over 4 heads + relu
  }
}

// ---------------------------------------------------------------------------
extern "C" void kernel_launch(void* const* d_in, const int* in_sizes, int n_in,
                              void* d_out, int out_size, void* d_ws, size_t ws_size,
                              hipStream_t stream) {
  (void)in_sizes; (void)n_in; (void)out_size; (void)ws_size;
  const float* nodes  = (const float*)d_in[0];
  const float* edges  = (const float*)d_in[1];
  const float* labels = (const float*)d_in[2];
  const float* We     = (const float*)d_in[3];
  const float* be     = (const float*)d_in[4];
  const float* W      = (const float*)d_in[5];
  const float* b      = (const float*)d_in[6];
  const float* u      = (const float*)d_in[7];
  const float* v      = (const float*)d_in[8];
  float* out = (float*)d_out;

  char* ws = (char*)d_ws;
  size_t off = 0;
  auto alloc = [&](size_t bytes) -> void* {
    void* p = ws + off;
    off += (bytes + 255) & ~(size_t)255;
    return p;
  };
  __bf16* nodes16 = (__bf16*)alloc((size_t)N_NODES * DMODEL * 2);            // 4 MB
  __bf16* el16    = (__bf16*)alloc((size_t)N_NODES * DMODEL * 2);            // 4 MB
  __bf16* WeT     = (__bf16*)alloc((size_t)DFF * DMODEL * 2);                // 0.5 MB
  __bf16* WT      = (__bf16*)alloc((size_t)NHEAD * DFF * DMODEL * 2);        // 2 MB
  __bf16* featsET = (__bf16*)alloc((size_t)NHEAD * DFF * N_NODES * 2);       // 16 MB
  float*  a_self  = (float*)alloc((size_t)NHEAD * N_NODES * 4);              // 64 KB
  float*  a_neigh = (float*)alloc((size_t)NHEAD * N_NODES * 4);              // 64 KB
  float*  stats   = (float*)alloc(2 * NHEAD * 4);

  const int totND = N_NODES * DMODEL;
  gat_prep_inputs<<<(totND + 255) / 256, 256, 0, stream>>>(nodes, edges, labels,
                                                           nodes16, el16, totND);
  const int totFD = DFF * DMODEL;
  gat_prep_weights<<<(totFD + 255) / 256, 256, 0, stream>>>(We, W, WeT, WT, totFD);

  // a_self / a_neigh are contiguous: one async memset (graph-capturable)
  hipMemsetAsync(a_self, 0, (size_t)2 * NHEAD * N_NODES * sizeof(float), stream);

  gat_proj_kernel<<<dim3(N_NODES / 32, DFF / 64), 256, 0, stream>>>(
      nodes16, el16, WeT, WT, be, b, u, v, featsET, a_self, a_neigh);

  gat_stats_kernel<<<NHEAD, 256, 0, stream>>>(a_neigh, stats);

  gat_attn_kernel<<<N_NODES / 32, 256, 0, stream>>>(a_self, a_neigh, stats,
                                                    featsET, out);
}
